// GCLSTM_63668595195950
// MI455X (gfx1250) — compile-verified, matrix operand order checked by reading, not run on hardware
//
#include <hip/hip_runtime.h>
#include <math.h>

typedef __attribute__((ext_vector_type(16))) _Float16 v16h;
typedef __attribute__((ext_vector_type(8)))  _Float16 v8h;
typedef __attribute__((ext_vector_type(8)))  float    v8f;

#define NNODE 50000
#define NEDGE 800000
#define D     128
#define KF    512   // fused K: [X | H | Tx1 | Tx2]

// ---------------- utility kernels ----------------

__global__ void fill_zero_f32(float* __restrict__ p, size_t n) {
    size_t i = (size_t)blockIdx.x * blockDim.x + threadIdx.x;
    size_t s = (size_t)gridDim.x * blockDim.x;
    for (; i < n; i += s) p[i] = 0.0f;
}

__global__ void f32_to_f16(const float* __restrict__ src, _Float16* __restrict__ dst, size_t n) {
    size_t i = (size_t)blockIdx.x * blockDim.x + threadIdx.x;
    size_t s = (size_t)gridDim.x * blockDim.x;
    for (; i < n; i += s) dst[i] = (_Float16)src[i];
}

// deg[row[e]] += w[e]
__global__ void deg_kernel(const long long* __restrict__ row,
                           const float* __restrict__ ew,
                           float* __restrict__ deg, int E) {
    int e = blockIdx.x * blockDim.x + threadIdx.x;
    if (e < E) atomicAdd(&deg[(int)row[e]], ew[e]);
}

// norm[e] = -dinv[row]*w*dinv[col]
__global__ void norm_kernel(const long long* __restrict__ row,
                            const long long* __restrict__ col,
                            const float* __restrict__ ew,
                            const float* __restrict__ deg,
                            float* __restrict__ norm, int E) {
    int e = blockIdx.x * blockDim.x + threadIdx.x;
    if (e >= E) return;
    float dr = deg[(int)row[e]], dc = deg[(int)col[e]];
    float ir = (dr > 0.0f) ? rsqrtf(dr) : 0.0f;
    float ic = (dc > 0.0f) ? rsqrtf(dc) : 0.0f;
    norm[e] = -ir * ew[e] * ic;
}

// dst[col[e], :] += norm[e] * src[row[e], :]  (32 lanes/edge, float4 per lane)
__global__ void scatter_kernel(const long long* __restrict__ row,
                               const long long* __restrict__ col,
                               const float* __restrict__ norm,
                               const float* __restrict__ src,
                               float* __restrict__ dst, int E) {
    long long t = (long long)blockIdx.x * blockDim.x + threadIdx.x;
    int e = (int)(t >> 5);
    if (e >= E) return;
    int q = ((int)t & 31) << 2;
    int r = (int)row[e], c = (int)col[e];
    float w = norm[e];
    const float4 v = *(const float4*)(src + (size_t)r * D + q);
    float* d = dst + (size_t)c * D + q;
    atomicAdd(d + 0, w * v.x);
    atomicAdd(d + 1, w * v.y);
    atomicAdd(d + 2, w * v.z);
    atomicAdd(d + 3, w * v.w);
}

// Tx2 = 2*S2 - H  (directly to f16 for the GEMM)
__global__ void tx2_fin_kernel(const float* __restrict__ S2, const float* __restrict__ H,
                               _Float16* __restrict__ T2_16, size_t n) {
    size_t i = (size_t)blockIdx.x * blockDim.x + threadIdx.x;
    size_t s = (size_t)gridDim.x * blockDim.x;
    for (; i < n; i += s) T2_16[i] = (_Float16)(2.0f * S2[i] - H[i]);
}

// Pack one gate's fused weight [W_g; cw_g0; cw_g1; cw_g2] (512x128) transposed
// into WT[n][k] f16 so B-fragment loads are contiguous; also fuse biases.
__global__ void pack_weights(const float* __restrict__ W,   // 128x128 (in,out)
                             const float* __restrict__ CW,  // 3x128x128 (hop,in,out)
                             const float* __restrict__ b,   // 128
                             const float* __restrict__ cb,  // 128
                             _Float16* __restrict__ WT,     // [128][512]
                             float* __restrict__ biasT) {   // [128]
    int idx = blockIdx.x * blockDim.x + threadIdx.x;
    if (idx < D) biasT[idx] = b[idx] + cb[idx];
    if (idx >= D * KF) return;
    int n = idx / KF, k = idx % KF;
    int seg = k >> 7, kk = k & 127;
    float v = (seg == 0) ? W[kk * D + n] : CW[(size_t)(seg - 1) * D * D + kk * D + n];
    WT[(size_t)n * KF + k] = (_Float16)v;
}

// ---------------- fused WMMA GEMM + LSTM epilogue ----------------

// hardware-transcendental forms: v_exp_f32 + v_rcp_f32 (TRANS pipe, co-executes with WMMA)
__device__ __forceinline__ float fast_sigmoid(float x) {
    return __builtin_amdgcn_rcpf(1.0f + __expf(-x));
}
__device__ __forceinline__ float fast_tanh(float x) {
    return 1.0f - 2.0f * __builtin_amdgcn_rcpf(1.0f + __expf(2.0f * x));
}

__global__ void __launch_bounds__(256)
gemm_lstm_kernel(const _Float16* __restrict__ X16,
                 const _Float16* __restrict__ H16,
                 const _Float16* __restrict__ T1,
                 const _Float16* __restrict__ T2,
                 const _Float16* __restrict__ WT,   // [4][128][512]
                 const float* __restrict__ biasT,   // [4][128]
                 const float* __restrict__ Cold,
                 float* __restrict__ Hnew,
                 float* __restrict__ Cnew) {
    const int lane  = threadIdx.x & 31;
    const int wave  = threadIdx.x >> 5;   // 0..7 -> output column tile
    const int row16 = lane & 15;
    const int half  = lane >> 4;
    const int m0 = blockIdx.x * 64;       // 64 rows per block (4 row sub-tiles)
    const int n0 = wave << 4;

    const _Float16* Asrc[4] = { X16, H16, T1, T2 };

    // clamped A-row index per row sub-tile (tail block reads row N-1; stores guarded)
    int arow[4];
    #pragma unroll
    for (int rt = 0; rt < 4; ++rt) {
        int r = m0 + rt * 16 + row16;
        arow[rt] = (r < NNODE) ? r : (NNODE - 1);
    }

    v8f acc[4][4] = {};                   // [gate i,f,c,o][row sub-tile]

    #pragma unroll
    for (int ks = 0; ks < 16; ++ks) {
        const _Float16* A = Asrc[ks >> 2];
        const int k0 = (ks & 3) << 5;

        // 4 A fragments (16x32 f16 layout: two contiguous 16B chunks per lane)
        v16h afrag[4];
        #pragma unroll
        for (int rt = 0; rt < 4; ++rt) {
            const _Float16* ap = A + (size_t)arow[rt] * D + k0 + (half << 3);
            v8h lo = *(const v8h*)(ap);
            v8h hi = *(const v8h*)(ap + 16);
            afrag[rt] = __builtin_shufflevector(lo, hi,
                0,1,2,3,4,5,6,7,8,9,10,11,12,13,14,15);
        }
        // 4 B fragments (one per gate, same column tile)
        v16h bfrag[4];
        #pragma unroll
        for (int g = 0; g < 4; ++g) {
            const _Float16* bp = WT + ((size_t)(g * D + n0 + row16) * KF)
                                    + (ks << 5) + (half << 3);
            v8h lo = *(const v8h*)(bp);
            v8h hi = *(const v8h*)(bp + 16);
            bfrag[g] = __builtin_shufflevector(lo, hi,
                0,1,2,3,4,5,6,7,8,9,10,11,12,13,14,15);
        }
        // 8 fragments feed 16 WMMAs (~32 flops/byte of cache traffic)
        #pragma unroll
        for (int g = 0; g < 4; ++g) {
            #pragma unroll
            for (int rt = 0; rt < 4; ++rt) {
                acc[g][rt] = __builtin_amdgcn_wmma_f32_16x16x32_f16(
                    false, afrag[rt], false, bfrag[g], (short)0, acc[g][rt],
                    false, false);
            }
        }
    }

    // --- LSTM epilogue on f32 accumulators (C/D layout: M=r+8*half, N=lane%16)
    const int ncol = n0 + row16;
    const float bi = biasT[0 * D + ncol];
    const float bf = biasT[1 * D + ncol];
    const float bc = biasT[2 * D + ncol];
    const float bo = biasT[3 * D + ncol];
    #pragma unroll
    for (int rt = 0; rt < 4; ++rt) {
        #pragma unroll
        for (int r = 0; r < 8; ++r) {
            const int m = m0 + rt * 16 + r + (half << 3);
            if (m < NNODE) {
                const size_t off = (size_t)m * D + ncol;
                float I = fast_sigmoid(acc[0][rt][r] + bi);
                float F = fast_sigmoid(acc[1][rt][r] + bf);
                float T = fast_tanh(acc[2][rt][r] + bc);
                float O = fast_sigmoid(acc[3][rt][r] + bo);
                float cn = F * Cold[off] + I * T;
                Hnew[off] = O * fast_tanh(cn);
                Cnew[off] = cn;
            }
        }
    }
}

// ---------------- host launcher ----------------

extern "C" void kernel_launch(void* const* d_in, const int* in_sizes, int n_in,
                              void* d_out, int out_size, void* d_ws, size_t ws_size,
                              hipStream_t stream) {
    (void)in_sizes; (void)n_in; (void)out_size; (void)ws_size;
    const float*     X   = (const float*)d_in[0];
    const long long* ei  = (const long long*)d_in[1];   // int64 (2,E)
    const float*     ew  = (const float*)d_in[2];
    const float*     H   = (const float*)d_in[3];
    const float*     C   = (const float*)d_in[4];
    // gate order i,f,c,o at indices 5.. : W, b, cheb_w, cheb_b
    const float* Wg[4]  = { (const float*)d_in[5],  (const float*)d_in[9],
                            (const float*)d_in[13], (const float*)d_in[17] };
    const float* bg[4]  = { (const float*)d_in[6],  (const float*)d_in[10],
                            (const float*)d_in[14], (const float*)d_in[18] };
    const float* CWg[4] = { (const float*)d_in[7],  (const float*)d_in[11],
                            (const float*)d_in[15], (const float*)d_in[19] };
    const float* cbg[4] = { (const float*)d_in[8],  (const float*)d_in[12],
                            (const float*)d_in[16], (const float*)d_in[20] };

    const long long* row = ei;
    const long long* col = ei + NEDGE;

    float* Hnew = (float*)d_out;
    float* Cnew = Hnew + (size_t)NNODE * D;

    // workspace carve-up (256B aligned)
    char* ws = (char*)d_ws;
    size_t off = 0;
    auto alloc = [&](size_t bytes) {
        off = (off + 255) & ~(size_t)255;
        size_t cur = off; off += bytes; return cur;
    };
    float*    deg   = (float*)   (ws + alloc((size_t)NNODE * 4));
    float*    norm  = (float*)   (ws + alloc((size_t)NEDGE * 4));
    float*    Tx1   = (float*)   (ws + alloc((size_t)NNODE * D * 4));
    float*    S2    = (float*)   (ws + alloc((size_t)NNODE * D * 4));
    _Float16* X16   = (_Float16*)(ws + alloc((size_t)NNODE * D * 2));
    _Float16* H16   = (_Float16*)(ws + alloc((size_t)NNODE * D * 2));
    _Float16* T116  = (_Float16*)(ws + alloc((size_t)NNODE * D * 2));
    _Float16* T216  = (_Float16*)(ws + alloc((size_t)NNODE * D * 2));
    _Float16* WT    = (_Float16*)(ws + alloc((size_t)4 * D * KF * 2));
    float*    biasT = (float*)   (ws + alloc((size_t)4 * D * 4));

    const size_t ND = (size_t)NNODE * D;
    const int eblk = (NEDGE + 255) / 256;
    const int sblk = (int)(((size_t)NEDGE * 32 + 255) / 256);

    // 1) degree + edge norm
    fill_zero_f32<<<256, 256, 0, stream>>>(deg, (size_t)NNODE);
    deg_kernel<<<eblk, 256, 0, stream>>>(row, ew, deg, NEDGE);
    norm_kernel<<<eblk, 256, 0, stream>>>(row, col, ew, deg, norm, NEDGE);

    // 2) zero scatter targets, cast dense operands
    fill_zero_f32<<<4096, 256, 0, stream>>>(Tx1, ND);
    fill_zero_f32<<<4096, 256, 0, stream>>>(S2, ND);
    f32_to_f16<<<4096, 256, 0, stream>>>(X, X16, ND);
    f32_to_f16<<<4096, 256, 0, stream>>>(H, H16, ND);

    // 3) Chebyshev recursion (gate-independent): Tx1 = scatter(norm*H), Tx2 = 2*scatter(norm*Tx1) - H
    scatter_kernel<<<sblk, 256, 0, stream>>>(row, col, norm, H, Tx1, NEDGE);
    f32_to_f16<<<4096, 256, 0, stream>>>(Tx1, T116, ND);
    scatter_kernel<<<sblk, 256, 0, stream>>>(row, col, norm, Tx1, S2, NEDGE);
    tx2_fin_kernel<<<4096, 256, 0, stream>>>(S2, H, T216, ND);

    // 4) pack fused transposed weights + biases per gate
    for (int g = 0; g < 4; ++g) {
        pack_weights<<<(D * KF + 255) / 256, 256, 0, stream>>>(
            Wg[g], CWg[g], bg[g], cbg[g],
            WT + (size_t)g * D * KF, biasT + (size_t)g * D);
    }

    // 5) fused WMMA GEMM (4 gates, 64-row M-tiles, K=512) + LSTM epilogue
    gemm_lstm_kernel<<<(NNODE + 63) / 64, 256, 0, stream>>>(
        X16, H16, T116, T216, WT, biasT, C, Hnew, Cnew);
}
